// MSDesGATnet_45655502356934
// MI455X (gfx1250) — matrix operand order repaired and checked by reading, not compiled, not run on hardware
//
#include <hip/hip_runtime.h>
#include <hip/hip_bf16.h>

// ---------------------------------------------------------------------------
// MSDesGAT pipeline for gfx1250 (MI455X, wave32, WMMA).
// Heavy GEMMs -> v_wmma_f32_16x16x32_bf16 with double-buffered LDS fed by
// GLOBAL_LOAD_ASYNC_TO_LDS_B128 (ASYNCcnt). Weights pre-transposed to [N][K]
// so both A and B fragments are contiguous ds_load_b128 reads.
// Irregular graph ops -> f32 atomics.
// ---------------------------------------------------------------------------

typedef __bf16 bf16;
typedef __attribute__((ext_vector_type(16))) __bf16 v16bf;
typedef __attribute__((ext_vector_type(8)))  float  v8f;

#define NPIX   65536      // 256*256
#define FFIN   1024
#define LDSP   40         // padded LDS row stride (elements): 80B = 5*16B, bank-conflict free

__device__ __forceinline__ float lrelu(float v, float s) { return v >= 0.f ? v : s * v; }

// ---- ordered-uint mapping for float atomicMax --------------------------------
__device__ __forceinline__ unsigned fkey(float f) {
    unsigned u = __float_as_uint(f);
    return (u & 0x80000000u) ? ~u : (u | 0x80000000u);
}
__device__ __forceinline__ float funkey(unsigned k) {
    return (k & 0x80000000u) ? __uint_as_float(k ^ 0x80000000u) : __uint_as_float(~k);
}

__device__ __forceinline__ void async_b128(unsigned lds_off, const void* gaddr) {
    asm volatile("global_load_async_to_lds_b128 %0, %1, off"
                 :: "v"(lds_off), "v"((unsigned long long)(size_t)gaddr)
                 : "memory");
}

// ===========================================================================
// bf16 WMMA GEMM: C[M,N] = act( A[M,K](bf16) * Bt[N,K](bf16)^T + bias[N] )
// Block = 256 threads (8 waves). Tile 128(M) x 128(N), K-step 32.
// Wave grid 4(M) x 2(N); each wave: 2 m-frags x 4 n-frags of 16x16.
// Double-buffered LDS, filled with async global->LDS b128 copies.
// ===========================================================================
__global__ __launch_bounds__(256)
void gemm_bf16_wmma(const bf16* __restrict__ A, const bf16* __restrict__ Bt,
                    const float* __restrict__ bias, float* __restrict__ C,
                    int M, int N, int K, int lda, int ldbt, int ldc,
                    float slope, int act)
{
    __shared__ bf16 As[2][128][LDSP];
    __shared__ bf16 Bs[2][128][LDSP];

    const int tid    = threadIdx.x;
    const int lane   = tid & 31;
    const int wid    = tid >> 5;
    const int wm     = (wid & 3) * 32;   // wave M offset in tile
    const int wn     = (wid >> 2) * 64;  // wave N offset in tile
    const int half   = lane >> 4;        // lane half (0/1)
    const int l16    = lane & 15;
    const int tile_m = blockIdx.y * 128;
    const int tile_n = blockIdx.x * 128;

    // per-thread copy slots: rows r0 and r0+64, 8 bf16 (16B) at column c0
    const int r0 = tid >> 2;
    const int c0 = (tid & 3) << 3;

    v8f acc[2][4];
#pragma unroll
    for (int i = 0; i < 2; ++i)
#pragma unroll
        for (int j = 0; j < 4; ++j)
#pragma unroll
            for (int e = 0; e < 8; ++e) acc[i][j][e] = 0.0f;

    auto issue = [&](int k0, int st) {
        const bf16* gA0 = &A [(size_t)(tile_m + r0)      * lda  + k0 + c0];
        const bf16* gA1 = &A [(size_t)(tile_m + r0 + 64) * lda  + k0 + c0];
        const bf16* gB0 = &Bt[(size_t)(tile_n + r0)      * ldbt + k0 + c0];
        const bf16* gB1 = &Bt[(size_t)(tile_n + r0 + 64) * ldbt + k0 + c0];
        async_b128((unsigned)(size_t)&As[st][r0][c0],      gA0);
        async_b128((unsigned)(size_t)&As[st][r0 + 64][c0], gA1);
        async_b128((unsigned)(size_t)&Bs[st][r0][c0],      gB0);
        async_b128((unsigned)(size_t)&Bs[st][r0 + 64][c0], gB1);
        __builtin_prefetch((const char*)gA0 + 64, 0, 0);   // speculative next K-tile
    };

    issue(0, 0);
    const int T = K >> 5;
    for (int i = 0; i < T; ++i) {
        const int st = i & 1;
        if (i + 1 < T) {
            issue((i + 1) << 5, st ^ 1);                    // overlap next tile with compute
            asm volatile("s_wait_asynccnt 0x4" ::: "memory");
        } else {
            asm volatile("s_wait_asynccnt 0x0" ::: "memory");
        }
        __syncthreads();

        // B fragments: contiguous in Bt-LDS -> 2x ds_load_b128 each
        v16bf bfr[4];
#pragma unroll
        for (int nf = 0; nf < 4; ++nf) {
            const int col = wn + nf * 16 + l16;
#pragma unroll
            for (int e = 0; e < 16; ++e) bfr[nf][e] = Bs[st][col][half * 16 + e];
        }
#pragma unroll
        for (int mf = 0; mf < 2; ++mf) {
            const int row = wm + mf * 16 + l16;
            v16bf afr;   // two contiguous 8-elem chunks -> 2x ds_load_b128
#pragma unroll
            for (int e = 0; e < 16; ++e) {
                int k = half * 16 + e;
                afr[e] = As[st][row][k];
            }
#pragma unroll
            for (int nf = 0; nf < 4; ++nf)
                acc[mf][nf] = __builtin_amdgcn_wmma_f32_16x16x32_bf16(
                    false, afr, false, bfr[nf], (short)0, acc[mf][nf], false, false);
        }
        __syncthreads();
    }

    // epilogue: D layout -> row = half*8 + r, col = l16 within each 16x16 frag
#pragma unroll
    for (int mf = 0; mf < 2; ++mf)
#pragma unroll
        for (int nf = 0; nf < 4; ++nf)
#pragma unroll
            for (int r = 0; r < 8; ++r) {
                int row = tile_m + wm + mf * 16 + half * 8 + r;
                int col = tile_n + wn + nf * 16 + l16;
                float v = acc[mf][nf][r];
                if (bias) v += bias[col];
                if (act)  v = lrelu(v, slope);
                C[(size_t)row * ldc + col] = v;
            }
}

// ===========================================================================
// Per-column (feature/channel) mean + inv-std over `rows` rows, pitch `ld`.
// ===========================================================================
__global__ __launch_bounds__(256)
void col_stats(const float* __restrict__ X, float* mean, float* istd, int rows, int ld)
{
    __shared__ float ssum[256], ssq[256];
    int c = blockIdx.x;
    float s = 0.f, q = 0.f;
    for (int r = threadIdx.x; r < rows; r += 256) {
        float v = X[(size_t)r * ld + c];
        s += v; q += v * v;
    }
    ssum[threadIdx.x] = s; ssq[threadIdx.x] = q;
    __syncthreads();
    for (int st = 128; st > 0; st >>= 1) {
        if (threadIdx.x < st) {
            ssum[threadIdx.x] += ssum[threadIdx.x + st];
            ssq[threadIdx.x]  += ssq[threadIdx.x + st];
        }
        __syncthreads();
    }
    if (threadIdx.x == 0) {
        float m = ssum[0] / rows;
        float v = ssq[0] / rows - m * m;
        mean[c] = m;
        istd[c] = rsqrtf(v + 1e-5f);
    }
}

__global__ void bn_norm_bf16(const float* __restrict__ X, const float* mean,
                             const float* istd, const float* g, const float* b,
                             bf16* __restrict__ out, long long total, int cols, int ld)
{
    long long i = (long long)blockIdx.x * blockDim.x + threadIdx.x;
    if (i >= total) return;
    int c = (int)(i % cols);
    long long r = i / cols;
    float v = (X[r * ld + c] - mean[c]) * istd[c] * g[c] + b[c];
    out[i] = (bf16)v;
}

// Produce Wt[N][K] bf16.  src_is_nk=1: src is row-major [N][K]; else [K][N].
__global__ void make_wt_bf16(const float* __restrict__ src, bf16* __restrict__ dst,
                             int K, int N, int src_is_nk)
{
    long long i = (long long)blockIdx.x * blockDim.x + threadIdx.x;
    if (i >= (long long)K * N) return;
    int n = (int)(i / K), k = (int)(i % K);
    dst[i] = (bf16)(src_is_nk ? src[(size_t)n * K + k] : src[(size_t)k * N + n]);
}

__global__ void cvt_bf16(const float* __restrict__ src, bf16* __restrict__ dst, long long n)
{
    long long i = (long long)blockIdx.x * blockDim.x + threadIdx.x;
    if (i < n) dst[i] = (bf16)src[i];
}

__global__ void zero_f32(float* p, long long n)
{
    long long i = (long long)blockIdx.x * blockDim.x + threadIdx.x;
    if (i < n) p[i] = 0.f;
}

// ===========================================================================
// Superpixel mean pooling (segment mean) via atomics.
// ===========================================================================
__global__ void pool_cnt(const int* __restrict__ sp, float* cnt)
{
    int p = blockIdx.x * blockDim.x + threadIdx.x;
    if (p < NPIX) atomicAdd(&cnt[sp[p]], 1.0f);
}

__global__ void pool_accum(const float* __restrict__ xf, const int* __restrict__ sp,
                           float* __restrict__ sum, int din, int ld)
{
    long long i = (long long)blockIdx.x * blockDim.x + threadIdx.x;
    if (i >= (long long)NPIX * din) return;
    int c = (int)(i % din);
    int p = (int)(i / din);
    atomicAdd(&sum[(size_t)sp[p] * din + c], xf[(size_t)p * ld + c]);
}

__global__ void pool_final(float* __restrict__ sum, const float* __restrict__ cnt,
                           long long total, int din)
{
    long long i = (long long)blockIdx.x * blockDim.x + threadIdx.x;
    if (i >= total) return;
    sum[i] /= fmaxf(cnt[i / din], 1.0f);
}

// ===========================================================================
// GATv2 edge ops (self loops appended logically: e >= E -> src = dst = e-E).
// ===========================================================================
__global__ void gat_score(const float* __restrict__ xl, const float* __restrict__ xr,
                          const float* __restrict__ att, const int* __restrict__ ei,
                          int E, int n, int heads, int doth,
                          float* __restrict__ score, unsigned* __restrict__ smax)
{
    long long t = (long long)blockIdx.x * blockDim.x + threadIdx.x;
    long long Etot = (long long)E + n;
    if (t >= Etot * heads) return;
    int e = (int)(t / heads), h = (int)(t % heads);
    int src = (e < E) ? ei[e]     : (e - E);
    int dst = (e < E) ? ei[E + e] : (e - E);
    int F = heads * doth;
    const float* pl = xl + (size_t)src * F + h * doth;
    const float* pr = xr + (size_t)dst * F + h * doth;
    const float* pa = att + (size_t)h * doth;
    float s = 0.f;
    for (int d = 0; d < doth; ++d) s += pa[d] * lrelu(pl[d] + pr[d], 0.2f);
    score[t] = s;
    atomicMax(&smax[(size_t)dst * heads + h], fkey(s));
}

__global__ void gat_exp(const float* __restrict__ score, const unsigned* __restrict__ smax,
                        const int* __restrict__ ei, int E, int n, int heads,
                        float* __restrict__ ex, float* __restrict__ denom)
{
    long long t = (long long)blockIdx.x * blockDim.x + threadIdx.x;
    long long Etot = (long long)E + n;
    if (t >= Etot * heads) return;
    int e = (int)(t / heads), h = (int)(t % heads);
    int dst = (e < E) ? ei[E + e] : (e - E);
    float v = expf(score[t] - funkey(smax[(size_t)dst * heads + h]));
    ex[t] = v;
    atomicAdd(&denom[(size_t)dst * heads + h], v);
}

__global__ void gat_agg(const float* __restrict__ xl, const float* __restrict__ ex,
                        const float* __restrict__ denom, const int* __restrict__ ei,
                        int E, int n, int heads, int doth, float* __restrict__ out)
{
    int F = heads * doth;
    long long t = (long long)blockIdx.x * blockDim.x + threadIdx.x;
    long long Etot = (long long)E + n;
    if (t >= Etot * F) return;
    int e = (int)(t / F), f = (int)(t % F), h = f / doth;
    int src = (e < E) ? ei[e]     : (e - E);
    int dst = (e < E) ? ei[E + e] : (e - E);
    float alpha = ex[(size_t)e * heads + h] / denom[(size_t)dst * heads + h];
    atomicAdd(&out[(size_t)dst * F + f], alpha * xl[(size_t)src * F + f]);
}

__global__ void bias_act(float* __restrict__ X, const float* __restrict__ bias,
                         long long total, int F, float slope, int act)
{
    long long i = (long long)blockIdx.x * blockDim.x + threadIdx.x;
    if (i >= total) return;
    float v = X[i] + bias[i % F];
    if (act) v = lrelu(v, slope);
    X[i] = v;
}

__global__ void unpool(const float* __restrict__ hg, const int* __restrict__ sp,
                       float* __restrict__ xf, int dout, int off)
{
    long long i = (long long)blockIdx.x * blockDim.x + threadIdx.x;
    if (i >= (long long)NPIX * dout) return;
    int p = (int)(i / dout);
    int c = (int)(i % dout);
    xf[(size_t)p * FFIN + off + c] = hg[(size_t)sp[p] * dout + c];
}

// ===========================================================================
// NHWC 5x5 depthwise conv, SAME padding, bias + LeakyReLU(0.01).
// ===========================================================================
__global__ __launch_bounds__(256)
void dwconv5x5(const float* __restrict__ in, const float* __restrict__ w,
               const float* __restrict__ bias, float* __restrict__ out,
               int Cw, int out_ld, int out_off)
{
    int p = blockIdx.x;
    int h = p >> 8;
    int x = p & 255;
    for (int c = threadIdx.x; c < Cw; c += 256) {
        float s = bias[c];
#pragma unroll
        for (int kh = 0; kh < 5; ++kh) {
            int hy = h + kh - 2;
            if ((unsigned)hy >= 256u) continue;
#pragma unroll
            for (int kw = 0; kw < 5; ++kw) {
                int wx = x + kw - 2;
                if ((unsigned)wx >= 256u) continue;
                s += in[((size_t)(hy << 8) + wx) * Cw + c] * w[c * 25 + kh * 5 + kw];
            }
        }
        out[(size_t)p * out_ld + out_off + c] = lrelu(s, 0.01f);
    }
}

// final linear 1024 -> 16 (bandwidth bound)
__global__ void linear16(const float* __restrict__ xf, const float* __restrict__ W,
                         const float* __restrict__ b, float* __restrict__ out)
{
    long long t = (long long)blockIdx.x * blockDim.x + threadIdx.x;
    if (t >= (long long)NPIX * 16) return;
    int p = (int)(t >> 4), j = (int)(t & 15);
    const float* row = xf + (size_t)p * FFIN;
    float s = b[j];
    for (int k = 0; k < FFIN; ++k) s += row[k] * W[k * 16 + j];
    out[t] = s;
}

// ===========================================================================
// Host-side orchestration
// ===========================================================================
static inline dim3 g1(long long n) { return dim3((unsigned)((n + 255) / 256)); }

extern "C" void kernel_launch(void* const* d_in, const int* in_sizes, int n_in,
                              void* d_out, int out_size, void* d_ws, size_t ws_size,
                              hipStream_t stream)
{
    (void)in_sizes; (void)n_in; (void)out_size; (void)ws_size;

    const float* x     = (const float*)d_in[0];
    const int*   sp[3] = {(const int*)d_in[1], (const int*)d_in[2], (const int*)d_in[3]};
    const int*   ei[3] = {(const int*)d_in[4], (const int*)d_in[5], (const int*)d_in[6]};
    const float* bn1_g = (const float*)d_in[7];
    const float* bn1_b = (const float*)d_in[8];
    const float* pw1_w = (const float*)d_in[9];
    const float* pw1_b = (const float*)d_in[10];
    const float* dw1_w = (const float*)d_in[11];
    const float* dw1_b = (const float*)d_in[12];
    const float* bn2_g = (const float*)d_in[13];
    const float* bn2_b = (const float*)d_in[14];
    const float* pw2_w = (const float*)d_in[15];
    const float* pw2_b = (const float*)d_in[16];
    const float* dw2_w = (const float*)d_in[17];
    const float* dw2_b = (const float*)d_in[18];
    const float* lin_w = (const float*)d_in[61];
    const float* lin_b = (const float*)d_in[62];
    auto GP = [&](int i, int j) -> const float* { return (const float*)d_in[19 + i * 14 + j]; };

    // ---- workspace carve ----
    char* wsp = (char*)d_ws;
    size_t off = 0;
    auto carve = [&](size_t bytes) -> void* {
        void* p = wsp + off;
        off = (off + bytes + 255) & ~(size_t)255;
        return p;
    };
    float*    xf    = (float*)carve((size_t)NPIX * FFIN * 4);      // feature bank (f32)
    float*    tmp   = (float*)carve((size_t)NPIX * FFIN * 4);      // pointwise outputs
    bf16*     xbf   = (bf16*)carve((size_t)NPIX * FFIN * 2);       // bf16 GEMM A
    bf16*     wbfA  = (bf16*)carve((size_t)FFIN * FFIN * 2);       // bf16 weights (Wt[N][K])
    bf16*     wbfB  = (bf16*)carve((size_t)FFIN * FFIN * 2);
    float*    mean  = (float*)carve(FFIN * 4);
    float*    istd  = (float*)carve(FFIN * 4);
    float*    spsum = (float*)carve((size_t)4096 * 128 * 4);       // max n*din
    float*    cnt   = (float*)carve(4096 * 4);
    bf16*     gbf   = (bf16*)carve((size_t)4096 * 256 * 2);        // bf16 node feats
    float*    xl    = (float*)carve((size_t)4096 * 256 * 4);
    float*    xr    = (float*)carve((size_t)4096 * 256 * 4);
    float*    score = (float*)carve((size_t)(32768 + 4096) * 4 * 4);
    float*    ex    = (float*)carve((size_t)(32768 + 4096) * 4 * 4);
    unsigned* smax  = (unsigned*)carve((size_t)4096 * 4 * 4);
    float*    denom = (float*)carve((size_t)4096 * 4 * 4);
    float*    agg   = (float*)carve((size_t)4096 * 256 * 4);       // GAT layer-1 out
    float*    hg    = (float*)carve((size_t)4096 * 128 * 4);       // GAT layer-2 out

    const int NSPc[3] = {4096, 1024, 256};
    const int NEc[3]  = {32768, 8192, 2048};
    const int DIN[3]  = {128, 256, 512};

    // ============== SSConv1 (128 -> 128) ==============
    col_stats<<<128, 256, 0, stream>>>(x, mean, istd, NPIX, 128);
    bn_norm_bf16<<<g1((long long)NPIX * 128), 256, 0, stream>>>(
        x, mean, istd, bn1_g, bn1_b, xbf, (long long)NPIX * 128, 128, 128);
    make_wt_bf16<<<g1(128 * 128), 256, 0, stream>>>(pw1_w, wbfA, 128, 128, 1); // (O,I) is [N][K]
    gemm_bf16_wmma<<<dim3(1, NPIX / 128), 256, 0, stream>>>(
        xbf, wbfA, pw1_b, tmp, NPIX, 128, 128, 128, 128, 128, 0.01f, 1);
    dwconv5x5<<<NPIX, 256, 0, stream>>>(tmp, dw1_w, dw1_b, xf, 128, FFIN, 0);

    // ============== 3 GNN scales ==============
    for (int i = 0; i < 3; ++i) {
        const int n = NSPc[i], E = NEc[i], din = DIN[i], dout = din;
        const long long Etot = (long long)E + n;

        // ---- superpixel mean pooling of current xf[:, :din] ----
        zero_f32<<<g1((long long)n * din), 256, 0, stream>>>(spsum, (long long)n * din);
        zero_f32<<<g1(n), 256, 0, stream>>>(cnt, n);
        pool_cnt<<<g1(NPIX), 256, 0, stream>>>(sp[i], cnt);
        pool_accum<<<g1((long long)NPIX * din), 256, 0, stream>>>(xf, sp[i], spsum, din, FFIN);
        pool_final<<<g1((long long)n * din), 256, 0, stream>>>(spsum, cnt, (long long)n * din, din);

        // ---- BN1d + bf16 ----
        col_stats<<<din, 256, 0, stream>>>(spsum, mean, istd, n, din);
        bn_norm_bf16<<<g1((long long)n * din), 256, 0, stream>>>(
            spsum, mean, istd, GP(i, 0), GP(i, 1), gbf, (long long)n * din, din, din);

        // ---- GATv2 layer 1 (din -> 4x64) ----
        make_wt_bf16<<<g1(din * 256), 256, 0, stream>>>(GP(i, 2), wbfA, din, 256, 0); // (K,N)
        gemm_bf16_wmma<<<dim3(2, n / 128), 256, 0, stream>>>(
            gbf, wbfA, GP(i, 3), xl, n, 256, din, din, din, 256, 0.f, 0);
        make_wt_bf16<<<g1(din * 256), 256, 0, stream>>>(GP(i, 4), wbfB, din, 256, 0);
        gemm_bf16_wmma<<<dim3(2, n / 128), 256, 0, stream>>>(
            gbf, wbfB, GP(i, 5), xr, n, 256, din, din, din, 256, 0.f, 0);

        zero_f32<<<g1((long long)n * 4), 256, 0, stream>>>((float*)smax, (long long)n * 4);
        gat_score<<<g1(Etot * 4), 256, 0, stream>>>(xl, xr, GP(i, 6), ei[i], E, n, 4, 64, score, smax);
        zero_f32<<<g1((long long)n * 4), 256, 0, stream>>>(denom, (long long)n * 4);
        gat_exp<<<g1(Etot * 4), 256, 0, stream>>>(score, smax, ei[i], E, n, 4, ex, denom);
        zero_f32<<<g1((long long)n * 256), 256, 0, stream>>>(agg, (long long)n * 256);
        gat_agg<<<g1(Etot * 256), 256, 0, stream>>>(xl, ex, denom, ei[i], E, n, 4, 64, agg);
        bias_act<<<g1((long long)n * 256), 256, 0, stream>>>(agg, GP(i, 7), (long long)n * 256, 256, 0.01f, 1);
        cvt_bf16<<<g1((long long)n * 256), 256, 0, stream>>>(agg, gbf, (long long)n * 256);

        // ---- GATv2 layer 2 (256 -> dout, 1 head) ----
        make_wt_bf16<<<g1(256 * dout), 256, 0, stream>>>(GP(i, 8), wbfA, 256, dout, 0);  // (K,N)
        gemm_bf16_wmma<<<dim3(dout / 128, n / 128), 256, 0, stream>>>(
            gbf, wbfA, GP(i, 9), xl, n, dout, 256, 256, 256, dout, 0.f, 0);
        make_wt_bf16<<<g1(256 * dout), 256, 0, stream>>>(GP(i, 10), wbfB, 256, dout, 0);
        gemm_bf16_wmma<<<dim3(dout / 128, n / 128), 256, 0, stream>>>(
            gbf, wbfB, GP(i, 11), xr, n, dout, 256, 256, 256, dout, 0.f, 0);

        zero_f32<<<g1(n), 256, 0, stream>>>((float*)smax, n);
        gat_score<<<g1(Etot), 256, 0, stream>>>(xl, xr, GP(i, 12), ei[i], E, n, 1, dout, score, smax);
        zero_f32<<<g1(n), 256, 0, stream>>>(denom, n);
        gat_exp<<<g1(Etot), 256, 0, stream>>>(score, smax, ei[i], E, n, 1, ex, denom);
        zero_f32<<<g1((long long)n * dout), 256, 0, stream>>>(hg, (long long)n * dout);
        gat_agg<<<g1(Etot * dout), 256, 0, stream>>>(xl, ex, denom, ei[i], E, n, 1, dout, hg);
        bias_act<<<g1((long long)n * dout), 256, 0, stream>>>(hg, GP(i, 13), (long long)n * dout, dout, 0.f, 0);

        // ---- unpool: append columns [din, 2*din) of xf ----
        unpool<<<g1((long long)NPIX * dout), 256, 0, stream>>>(hg, sp[i], xf, dout, din);
    }

    // ============== SSConv2 (1024 -> 1024) — the 137 GFLOP GEMM ==============
    col_stats<<<FFIN, 256, 0, stream>>>(xf, mean, istd, NPIX, FFIN);
    bn_norm_bf16<<<g1((long long)NPIX * FFIN), 256, 0, stream>>>(
        xf, mean, istd, bn2_g, bn2_b, xbf, (long long)NPIX * FFIN, FFIN, FFIN);
    make_wt_bf16<<<g1((long long)FFIN * FFIN), 256, 0, stream>>>(pw2_w, wbfA, FFIN, FFIN, 1); // (O,I)
    gemm_bf16_wmma<<<dim3(FFIN / 128, NPIX / 128), 256, 0, stream>>>(
        xbf, wbfA, pw2_b, tmp, NPIX, FFIN, FFIN, FFIN, FFIN, FFIN, 0.01f, 1);
    dwconv5x5<<<NPIX, 256, 0, stream>>>(tmp, dw2_w, dw2_b, xf, FFIN, FFIN, 0);

    // ============== final linear 1024 -> 16 ==============
    linear16<<<g1((long long)NPIX * 16), 256, 0, stream>>>(xf, lin_w, lin_b, (float*)d_out);
}
